// PhysicalWeaver_40553081209137
// MI455X (gfx1250) — matrix operand order repaired and checked by reading, not compiled, not run on hardware
//
#include <hip/hip_runtime.h>
#include <hip/hip_bf16.h>

typedef __attribute__((ext_vector_type(16))) _Float16 v16h;
typedef __attribute__((ext_vector_type(8)))  _Float16 v8h;
typedef __attribute__((ext_vector_type(8)))  float    v8f;

#define B_  4
#define N_  4096
#define INF_ 512
#define OUTF_ 512
#define DH 64
#define NQ 64          // query rows per block
#define NK 64          // keys per chunk
#define PADH 72        // padded half-stride for 64-wide f16 rows (144B, 16B aligned, conflict-free)
#define PADP 40        // padded half-stride for 32-wide f16 rows (80B, 16B aligned)

#if defined(__gfx1250__) && __has_builtin(__builtin_amdgcn_global_load_async_to_lds_b128) && \
    __has_builtin(__builtin_amdgcn_s_wait_asynccnt)
#define USE_ASYNC 1
#else
#define USE_ASYNC 0
#endif

__device__ __forceinline__ v16h ld16(const _Float16* lo, const _Float16* hi) {
    v8h a = *(const v8h*)lo;
    v8h b = *(const v8h*)hi;
    return __builtin_shufflevector(a, b, 0,1,2,3,4,5,6,7,8,9,10,11,12,13,14,15);
}

#if USE_ASYNC
typedef __attribute__((__vector_size__(16))) int cp_i4;
typedef __attribute__((address_space(1))) cp_i4* g_i4p;   // global (AS1)
typedef __attribute__((address_space(3))) cp_i4* l_i4p;   // LDS (AS3, 32-bit)
#endif

// 16B global -> LDS copy; async (ASYNCcnt, no VGPR staging) when available.
__device__ __forceinline__ void cp16(const _Float16* g, _Float16* l) {
#if USE_ASYNC
    // AS1 generic pointers are numerically identical; generic LDS pointer's low
    // 32 bits are the LDS offset.
    __builtin_amdgcn_global_load_async_to_lds_b128(
        (g_i4p)(unsigned long long)(__SIZE_TYPE__)g,
        (l_i4p)(unsigned int)(__SIZE_TYPE__)l, 0, 0);
#else
    *(v8h*)l = *(const v8h*)g;
#endif
}

__device__ __forceinline__ void wait_async() {
#if USE_ASYNC
    __builtin_amdgcn_s_wait_asynccnt(0);
#endif
}

// ---------------------------------------------------------------------------
// Kernel 1: fused projection  out[t, f] = sum_k x[t,k] * W[f,k],  f in [0,640)
//   f<64 -> Q (row-major, token x 64, f16)
//   f<128 -> K (row-major, token x 64, f16)
//   else  -> V stored TRANSPOSED per batch: vt[b][f-128][token] (f16)
// ---------------------------------------------------------------------------
__global__ __launch_bounds__(256)
void weaver_proj(const float* __restrict__ x,  const float* __restrict__ Wq,
                 const float* __restrict__ Wk, const float* __restrict__ Wv,
                 _Float16* __restrict__ qg, _Float16* __restrict__ kg,
                 _Float16* __restrict__ vtg)
{
    __shared__ _Float16 xs[64 * PADP];     // 64 tokens x 32 k
    __shared__ _Float16 wsh[128 * PADP];   // 128 feats x 32 k

    const int tid  = threadIdx.x;
    const int lane = tid & 31;
    const int w    = tid >> 5;      // 8 waves
    const int wr   = w & 3;         // token 16-group
    const int wh   = w >> 2;        // feat half (0/1)
    const int mbase = blockIdx.x * 64;     // token base, 0..16320
    const int fbase = blockIdx.y * 128;    // 0,128,256,384,512

    v8f acc[4];
    #pragma unroll
    for (int tf = 0; tf < 4; ++tf)
        #pragma unroll
        for (int v = 0; v < 8; ++v) acc[tf][v] = 0.f;

    for (int ks = 0; ks < INF_; ks += 32) {
        {   // stage x tile 64x32 (f32 -> f16)
            int row = tid >> 2, c = (tid & 3) * 8;
            const float* src = x + (size_t)(mbase + row) * INF_ + ks + c;
            #pragma unroll
            for (int j = 0; j < 8; ++j) xs[row * PADP + c + j] = (_Float16)src[j];
        }
        {   // stage W tile 128x32
            int row = tid >> 1, c = (tid & 1) * 16;
            int f = fbase + row;
            const float* wrow = (f < 64)  ? (Wq + (size_t)f * INF_)
                              : (f < 128) ? (Wk + (size_t)(f - 64) * INF_)
                                          : (Wv + (size_t)(f - 128) * INF_);
            #pragma unroll
            for (int j = 0; j < 16; ++j) wsh[row * PADP + c + j] = (_Float16)wrow[ks + c + j];
        }
        __syncthreads();

        // A fragment: 16 rows x 32 k, lane = row (lane&15), K interleaved per ISA layout
        const _Float16* ap = &xs[(16 * wr + (lane & 15)) * PADP + ((lane >= 16) ? 8 : 0)];
        v16h a = ld16(ap, ap + 16);
        #pragma unroll
        for (int tf = 0; tf < 4; ++tf) {
            const _Float16* bp = &wsh[(64 * wh + 16 * tf + (lane & 15)) * PADP + ((lane >= 16) ? 8 : 0)];
            v16h bb = ld16(bp, bp + 16);
            acc[tf] = __builtin_amdgcn_wmma_f32_16x16x32_f16(false, a, false, bb,
                                                             (short)0, acc[tf], false, false);
        }
        __syncthreads();
    }

    // D layout: lane holds col (lane&15); VGPR v -> row v (+8 for upper lanes)
    const int t0 = mbase + 16 * wr + ((lane >= 16) ? 8 : 0);
    #pragma unroll
    for (int tf = 0; tf < 4; ++tf) {
        int f = fbase + 64 * wh + 16 * tf + (lane & 15);
        if (f < 64) {
            #pragma unroll
            for (int v = 0; v < 8; ++v) qg[(size_t)(t0 + v) * DH + f] = (_Float16)acc[tf][v];
        } else if (f < 128) {
            #pragma unroll
            for (int v = 0; v < 8; ++v) kg[(size_t)(t0 + v) * DH + (f - 64)] = (_Float16)acc[tf][v];
        } else {
            int bat = t0 >> 12, tt = t0 & (N_ - 1);
            v8h pk;
            #pragma unroll
            for (int v = 0; v < 8; ++v) pk[v] = (_Float16)acc[tf][v];
            *(v8h*)(vtg + ((size_t)(bat * OUTF_ + (f - 128))) * N_ + tt) = pk;  // 8 consecutive tokens
        }
    }
}

// ---------------------------------------------------------------------------
// Kernel 2: flash attention with exact distance-softmax
//   logits = -max(|q|^2 + |k|^2 - 2 q.k, 0) / (2 lam^2), online softmax, O = P.V
// Block: 512 threads (16 waves, 4x4). Each block: 64 query rows x 512 features.
// K/V chunks double-buffered in LDS via GLOBAL_LOAD_ASYNC_TO_LDS (ASYNCcnt).
// ---------------------------------------------------------------------------
__global__ __launch_bounds__(512)
void weaver_attn(const _Float16* __restrict__ qg, const _Float16* __restrict__ kg,
                 const _Float16* __restrict__ vtg, const float* __restrict__ log_lambda,
                 float* __restrict__ out)
{
    __shared__ _Float16 Qs[NQ * PADH];              //  9216 B
    __shared__ _Float16 Ks[2][NK * PADH];           // 18432 B (double buffered)
    __shared__ _Float16 Vs[2][OUTF_ * PADH];        // 147456 B (feat-major, double buffered)
    __shared__ float    Ss[NQ * NK];                // 16384 B
    __shared__ _Float16 Ps[NQ * PADH];              //  9216 B
    __shared__ float qsqs[NQ], ksqs[NK], m_s[NQ], l_s[NQ], alpha_s[NQ], red[NQ * 4];

    const int tid  = threadIdx.x;
    const int lane = tid & 31;
    const int w    = tid >> 5;   // 0..15
    const int wr   = w >> 2, wc = w & 3;
    const int b    = blockIdx.y;
    const int qbase = blockIdx.x * NQ;

    const float lam  = fmaxf(expf(log_lambda[0]), 1.0e-3f);
    const float inv2 = 1.0f / (2.0f * lam * lam);

    const int krow = tid >> 3, kc8 = tid & 7;       // K staging assignment (512 x 16B)

    {   // load Q block: 64 x 64 halves = 512 x 16B
        *(v8h*)(&Qs[krow * PADH + kc8 * 8]) =
            *(const v8h*)(qg + (size_t)(b * N_ + qbase + krow) * DH + kc8 * 8);
    }
    if (tid < NQ) { m_s[tid] = -3.0e38f; l_s[tid] = 0.f; }

    // prefetch chunk 0 into buffer 0 (async: no VGPR staging, tracked by ASYNCcnt)
    cp16(kg + (size_t)(b * N_ + krow) * DH + kc8 * 8, &Ks[0][krow * PADH + kc8 * 8]);
    #pragma unroll
    for (int i = 0; i < 8; ++i) {
        int idx = tid + i * 512, fr = idx >> 3, f8 = idx & 7;
        cp16(vtg + ((size_t)(b * OUTF_ + fr)) * N_ + f8 * 8, &Vs[0][fr * PADH + f8 * 8]);
    }

    __syncthreads();
    if (tid < NQ) {
        float s = 0.f;
        for (int d = 0; d < DH; ++d) { float q = (float)Qs[tid * PADH + d]; s += q * q; }
        qsqs[tid] = s;
    }

    v8f accO[8];
    #pragma unroll
    for (int tf = 0; tf < 8; ++tf)
        #pragma unroll
        for (int v = 0; v < 8; ++v) accO[tf][v] = 0.f;

    for (int ch = 0; ch < N_ / NK; ++ch) {
        const int bs = ch & 1;
        wait_async();
        __syncthreads();            // chunk ch visible in Ks[bs], Vs[bs]

        // prefetch chunk ch+1 into the other buffer while we compute
        if (ch + 1 < N_ / NK) {
            const int nb = bs ^ 1;
            cp16(kg + (size_t)(b * N_ + (ch + 1) * NK + krow) * DH + kc8 * 8,
                 &Ks[nb][krow * PADH + kc8 * 8]);
            #pragma unroll
            for (int i = 0; i < 8; ++i) {
                int idx = tid + i * 512, fr = idx >> 3, f8 = idx & 7;
                cp16(vtg + ((size_t)(b * OUTF_ + fr)) * N_ + (ch + 1) * NK + f8 * 8,
                     &Vs[nb][fr * PADH + f8 * 8]);
            }
        }

        // S = Q.K^T  (16x16 tile per wave)
        v8f accS;
        #pragma unroll
        for (int v = 0; v < 8; ++v) accS[v] = 0.f;
        #pragma unroll
        for (int kk = 0; kk < 2; ++kk) {
            const _Float16* ap = &Qs[(16 * wr + (lane & 15)) * PADH + kk * 32 + ((lane >= 16) ? 8 : 0)];
            v16h a = ld16(ap, ap + 16);
            const _Float16* bp = &Ks[bs][(16 * wc + (lane & 15)) * PADH + kk * 32 + ((lane >= 16) ? 8 : 0)];
            v16h bb = ld16(bp, bp + 16);
            accS = __builtin_amdgcn_wmma_f32_16x16x32_f16(false, a, false, bb,
                                                          (short)0, accS, false, false);
        }
        // |k|^2 for this chunk (runs alongside WMMA waves)
        if (tid < NK) {
            float s = 0.f;
            for (int d = 0; d < DH; ++d) { float k = (float)Ks[bs][tid * PADH + d]; s += k * k; }
            ksqs[tid] = s;
        }
        __syncthreads();

        // distance -> logits, write S tile
        #pragma unroll
        for (int v = 0; v < 8; ++v) {
            int row = 16 * wr + v + ((lane >= 16) ? 8 : 0);
            int col = 16 * wc + (lane & 15);
            float d2 = fmaxf(qsqs[row] + ksqs[col] - 2.0f * accS[v], 0.0f);
            Ss[row * NK + col] = -d2 * inv2;
        }
        __syncthreads();

        // online softmax: 4 threads per row
        if (tid < 256) {
            int row = tid >> 2, q = tid & 3;
            float mx = -3.0e38f;
            #pragma unroll
            for (int j = 0; j < 16; ++j) mx = fmaxf(mx, Ss[row * NK + q * 16 + j]);
            red[row * 4 + q] = mx;
        }
        __syncthreads();
        if (tid < NQ) {
            float mnew = fmaxf(fmaxf(red[tid * 4], red[tid * 4 + 1]),
                               fmaxf(red[tid * 4 + 2], red[tid * 4 + 3]));
            mnew = fmaxf(mnew, m_s[tid]);
            alpha_s[tid] = __expf(m_s[tid] - mnew);
            m_s[tid] = mnew;
        }
        __syncthreads();
        if (tid < 256) {
            int row = tid >> 2, q = tid & 3;
            float mnew = m_s[row], s = 0.f;
            v8h p0, p1;
            #pragma unroll
            for (int j = 0; j < 8; ++j) {
                float p = __expf(Ss[row * NK + q * 16 + j] - mnew);
                s += p; p0[j] = (_Float16)p;
            }
            #pragma unroll
            for (int j = 0; j < 8; ++j) {
                float p = __expf(Ss[row * NK + q * 16 + 8 + j] - mnew);
                s += p; p1[j] = (_Float16)p;
            }
            *(v8h*)(&Ps[row * PADH + q * 16])     = p0;
            *(v8h*)(&Ps[row * PADH + q * 16 + 8]) = p1;
            red[row * 4 + q] = s;
        }
        __syncthreads();
        if (tid < NQ)
            l_s[tid] = l_s[tid] * alpha_s[tid] +
                       (red[tid * 4] + red[tid * 4 + 1] + red[tid * 4 + 2] + red[tid * 4 + 3]);

        // rescale O and accumulate P.V
        #pragma unroll
        for (int v = 0; v < 8; ++v) {
            int row = 16 * wr + v + ((lane >= 16) ? 8 : 0);
            float a = alpha_s[row];
            #pragma unroll
            for (int tf = 0; tf < 8; ++tf) accO[tf][v] *= a;
        }
        v16h a0, a1;
        {
            const _Float16* ap = &Ps[(16 * wr + (lane & 15)) * PADH + ((lane >= 16) ? 8 : 0)];
            a0 = ld16(ap, ap + 16);
            a1 = ld16(ap + 32, ap + 48);
        }
        #pragma unroll
        for (int tf = 0; tf < 8; ++tf) {
            int f = wc * 128 + tf * 16 + (lane & 15);
            const _Float16* bp = &Vs[bs][f * PADH + ((lane >= 16) ? 8 : 0)];
            v16h b0 = ld16(bp, bp + 16);
            v16h b1 = ld16(bp + 32, bp + 48);
            accO[tf] = __builtin_amdgcn_wmma_f32_16x16x32_f16(false, a0, false, b0,
                                                              (short)0, accO[tf], false, false);
            accO[tf] = __builtin_amdgcn_wmma_f32_16x16x32_f16(false, a1, false, b1,
                                                              (short)0, accO[tf], false, false);
        }
        __syncthreads();   // protect Ss/Ps reuse; Vs[bs]/Ks[bs] read complete
    }

    // epilogue: divide by row sums, store f32 output
    #pragma unroll
    for (int v = 0; v < 8; ++v) {
        int row = 16 * wr + v + ((lane >= 16) ? 8 : 0);
        float linv = 1.0f / l_s[row];
        float* op = out + ((size_t)(b * N_ + qbase + row)) * OUTF_ + wc * 128 + (lane & 15);
        #pragma unroll
        for (int tf = 0; tf < 8; ++tf) op[tf * 16] = accO[tf][v] * linv;
    }
}

// ---------------------------------------------------------------------------
extern "C" void kernel_launch(void* const* d_in, const int* in_sizes, int n_in,
                              void* d_out, int out_size, void* d_ws, size_t ws_size,
                              hipStream_t stream) {
    const float* x  = (const float*)d_in[0];
    const float* Wq = (const float*)d_in[1];
    const float* Wk = (const float*)d_in[2];
    const float* Wv = (const float*)d_in[3];
    const float* ll = (const float*)d_in[4];
    float* out = (float*)d_out;

    char* ws = (char*)d_ws;
    const size_t qkBytes = (size_t)B_ * N_ * DH * sizeof(_Float16);   // 2 MB each
    _Float16* qg  = (_Float16*)ws;
    _Float16* kg  = (_Float16*)(ws + qkBytes);
    _Float16* vtg = (_Float16*)(ws + 2 * qkBytes);                    // 16 MB (transposed V)

    weaver_proj<<<dim3((B_ * N_) / 64, (DH + DH + OUTF_) / 128), 256, 0, stream>>>(
        x, Wq, Wk, Wv, qg, kg, vtg);
    weaver_attn<<<dim3(N_ / NQ, B_), 512, 0, stream>>>(qg, kg, vtg, ll, out);
}
